// hCSSM_74560632259189
// MI455X (gfx1250) — compile-verified, matrix operand order checked by reading, not run on hardware
//
#include <hip/hip_runtime.h>
#include <hip/hip_bf16.h>

typedef __attribute__((ext_vector_type(2)))  float    v2f;
typedef __attribute__((ext_vector_type(8)))  float    v8f;
typedef __attribute__((ext_vector_type(16))) _Float16 v16h;

#define Bq 4
#define Tq 32
#define Hq 128
#define Wq 128
#define Cq 3
#define Wf 65
#define BT 128           // B*T
#define NEGC (-1e30f)

// ---- workspace layout (float offsets) ----
#define WS_COS 0                         // 128*128
#define WS_SIN 16384                     // 128*128
#define WS_FB  32768                     // 128*130  forward row-DFT B matrix
#define WS_IB  49408                     // 130*128  inverse row-DFT B matrix (weights folded)
#define WS_KE  66048                     // 128*65*3 complex
#define WS_KI  115968
#define WS_CTX 165888                    // 128*32 ctx sums
#define WS_GAT 169984                    // 4 * 128*3 gates
#define WS_S1  171520                    // BT*H*Wf*C complex = 6,389,760 floats
#define WS_U   6561280                   // same size

// ---------------- WMMA wrappers ----------------
__device__ inline v8f wmma_f32x4(v2f a, v2f b, v8f c) {
  // D = A(16x4,f32) * B(4x16,f32) + C
  return __builtin_amdgcn_wmma_f32_16x16x4_f32(false, a, false, b, (short)0, c, false, false);
}
__device__ inline v8f wmma_f16x32(v16h a, v16h b, v8f c) {
  return __builtin_amdgcn_wmma_f32_16x16x32_f16(false, a, false, b, (short)0, c, false, false);
}

// branchless tanh-approx gelu: 0.5x(1+tanh z) = x*sigmoid(2z)
__device__ inline float gelu_f(float x) {
  float z = 0.7978845608f * (x + 0.044715f * x * x * x);
  return x * __builtin_amdgcn_rcpf(1.0f + __expf(-2.0f * z));
}

// ---------------- tables ----------------
__global__ void k_tables(float* __restrict__ ws) {
  int i = blockIdx.x * blockDim.x + threadIdx.x;
  const float TWO_PI = 6.28318530717958647692f;
  if (i < 128 * 128) {
    int k = i >> 7, n = i & 127;
    float ang = TWO_PI * (float)((k * n) & 127) * (1.0f / 128.0f);
    ws[WS_COS + i] = __cosf(ang);
    ws[WS_SIN + i] = __sinf(ang);
  }
  if (i < 128 * 130) {           // FB[kk][n], n = 2k(+1): (cos, -sin) of 2*pi*k*kk/128
    int kk = i / 130, n = i % 130;
    int k = n >> 1;
    float ang = TWO_PI * (float)((k * kk) & 127) * (1.0f / 128.0f);
    ws[WS_FB + i] = (n & 1) ? -__sinf(ang) : __cosf(ang);
  }
  if (i < 130 * 128) {           // IB[kidx][w] with hermitian weights and 1/128
    int kidx = i >> 7, w = i & 127;
    int k = kidx >> 1;
    float wgt = (k == 0 || k == 64) ? 1.0f : 2.0f;
    float ang = TWO_PI * (float)((k * w) & 127) * (1.0f / 128.0f);
    float v = (kidx & 1) ? -__sinf(ang) : __cosf(ang);
    ws[WS_IB + i] = v * wgt * (1.0f / 128.0f);
  }
}

// ---------------- KE / KI spectra (direct DFT of padded 16x16 kernels) ----------------
__global__ void k_kspec(const float* __restrict__ k_exc, const float* __restrict__ k_inh,
                        float* __restrict__ ws) {
  int i = blockIdx.x * blockDim.x + threadIdx.x;       // H*Wf*C = 24960
  if (i >= Hq * Wf * Cq) return;
  int c = i % Cq, wf = (i / Cq) % Wf, h = i / (Cq * Wf);
  const float* COS = ws + WS_COS + 128;   // row k=1: cos(2*pi*m/128)
  const float* SIN = ws + WS_SIN + 128;
  float er = 0.f, ei = 0.f, gr = 0.f, gi = 0.f;
  for (int y = 0; y < 16; ++y) {
    for (int x = 0; x < 16; ++x) {
      int m = (h * (y + 56) + wf * (x + 56)) & 127;
      float cc = COS[m], ss = SIN[m];
      float ke = k_exc[(y * 16 + x) * Cq + c];
      float ki = k_inh[(y * 16 + x) * Cq + c];
      er += ke * cc; ei -= ke * ss;
      gr += ki * cc; gi -= ki * ss;
    }
  }
  ws[WS_KE + 2 * i] = er; ws[WS_KE + 2 * i + 1] = ei;
  ws[WS_KI + 2 * i] = gr; ws[WS_KI + 2 * i + 1] = gi;
}

// ---------------- conv(3x3,s2,SAME) + gelu + spatial sum : implicit GEMM via f16 WMMA ----
__global__ __launch_bounds__(256) void k_conv(const float* __restrict__ in,
                                              const float* __restrict__ w,
                                              const float* __restrict__ bias,
                                              float* __restrict__ ctx_sum) {
  // weight fragments pre-swizzled to per-lane order: 16 contiguous halves per lane
  __shared__ __align__(32) _Float16 Bfrag[2 * 32 * 16];   // [nt][lane][h]
  __shared__ float red[32];
  int f = blockIdx.x;                 // frame 0..127
  int tid = threadIdx.x;
  for (int idx = tid; idx < 2 * 32 * 16; idx += 256) {
    int h2 = idx & 15, ln = (idx >> 4) & 31, nt = idx >> 9;
    int K = (h2 < 8 ? h2 : h2 + 8) + (ln >> 4) * 8;       // ISA 16-bit B layout
    int n = nt * 16 + (ln & 15);
    Bfrag[idx] = (K < 27) ? (_Float16)w[K * 32 + n] : (_Float16)0.f;
  }
  if (tid < 32) red[tid] = 0.f;
  __syncthreads();

  int wave = tid >> 5, lane = tid & 31;
  int m = lane & 15, hi = lane >> 4;
  int p = (blockIdx.y * 8 + wave) * 16 + m;      // output pixel 0..4095
  int oy = p >> 6, ox = p & 63;
  const float* inf = in + (size_t)f * Hq * Wq * Cq;

  v16h a;
  for (int h2 = 0; h2 < 16; ++h2) {
    int K = (h2 < 8 ? h2 : h2 + 8) + hi * 8;     // ISA 16-bit A layout
    float v = 0.f;
    if (K < 27) {
      int ky = K / 9, r = K % 9, kx = r / 3, ci = r % 3;
      int iy = 2 * oy + ky, ix = 2 * ox + kx;    // SAME: pad_lo = 0
      if (iy < Hq && ix < Wq) v = inf[(iy * Wq + ix) * Cq + ci];
    }
    a[h2] = (_Float16)v;
  }

  float partial[2];
  for (int nt = 0; nt < 2; ++nt) {
    v16h bm = *reinterpret_cast<const v16h*>(&Bfrag[(nt * 32 + lane) * 16]); // 2x ds_load_b128
    v8f acc = {};
    acc = wmma_f16x32(a, bm, acc);
    int n = nt * 16 + (lane & 15);
    float s = 0.f, bv = bias[n];
    for (int r = 0; r < 8; ++r) s += gelu_f(acc[r] + bv);
    partial[nt] = s;
  }
  for (int nt = 0; nt < 2; ++nt) atomicAdd(&red[nt * 16 + (lane & 15)], partial[nt]);
  __syncthreads();
  if (tid < 32) atomicAdd(&ctx_sum[f * 32 + tid], red[tid]);
}

// ---------------- gates ----------------
__global__ void k_gates(const float* __restrict__ ws_ctx,
                        const float* __restrict__ wa, const float* __restrict__ ba,
                        const float* __restrict__ wd, const float* __restrict__ bd,
                        const float* __restrict__ wm, const float* __restrict__ bm,
                        const float* __restrict__ wg, const float* __restrict__ bg,
                        float* __restrict__ gates) {
  int i = blockIdx.x * blockDim.x + threadIdx.x;       // f*3+c
  if (i >= BT * Cq) return;
  int f = i / Cq, c = i % Cq;
  float sa = ba[c], sd = bd[c], sm = bm[c], sg = bg[c];
  for (int j = 0; j < 32; ++j) {
    float cx = ws_ctx[f * 32 + j] * (1.0f / 4096.0f);
    sa += cx * wa[j * Cq + c]; sd += cx * wd[j * Cq + c];
    sm += cx * wm[j * Cq + c]; sg += cx * wg[j * Cq + c];
  }
  gates[0 * BT * Cq + i] = __builtin_amdgcn_rcpf(1.f + __expf(-sa));
  gates[1 * BT * Cq + i] = __builtin_amdgcn_rcpf(1.f + __expf(-sd));
  gates[2 * BT * Cq + i] = (sm > 20.f) ? sm : log1pf(__expf(sm));
  gates[3 * BT * Cq + i] = (sg > 20.f) ? sg : log1pf(__expf(sg));
}

// ---------------- forward row rDFT as GEMM (f32 WMMA) ----------------
__global__ __launch_bounds__(32) void k_dft_row_fwd(const float* __restrict__ in,
                                                    const float* __restrict__ FB,
                                                    float* __restrict__ out) {
  // rows r = (f*H + h)*C + c, M = 49152; N = 130 (re/im interleaved); K = 128
  int mt = blockIdx.x, nt = blockIdx.y, lane = threadIdx.x;
  int m = lane & 15, hi = lane >> 4;
  int r = mt * 16 + m;
  int c = r % Cq, h = (r / Cq) & (Hq - 1), f = r / (Cq * Hq);
  const float* row = in + ((size_t)(f * Hq + h) * Wq) * Cq + c;   // stride C
  int nn = nt * 16 + (lane & 15);
  bool nok = nn < 130;
  v8f acc = {};
  for (int kk = 0; kk < 128; kk += 4) {
    int k0 = kk + hi * 2;
    v2f a = { row[k0 * Cq], row[(k0 + 1) * Cq] };
    v2f b = { nok ? FB[k0 * 130 + nn] : 0.f,
              nok ? FB[(k0 + 1) * 130 + nn] : 0.f };
    acc = wmma_f32x4(a, b, acc);
  }
  if (nok) {
    int k = nn >> 1, comp = nn & 1;
    for (int vr = 0; vr < 8; ++vr) {
      int rr = mt * 16 + vr + 8 * hi;
      int cc = rr % Cq, hh = (rr / Cq) & (Hq - 1), ff = rr / (Cq * Hq);
      out[((((size_t)ff * Hq + hh) * Wf + k) * Cq + cc) * 2 + comp] = acc[vr];
    }
  }
}

// ------- column complex DFT (fwd/inv): 4 real GEMMs, 2 n-tiles per block (f32 WMMA) ------
__global__ __launch_bounds__(32) void k_dft_col(const float* __restrict__ X,
                                                const float* __restrict__ ws,
                                                float* __restrict__ Y,
                                                float sgn, float scale) {
  // rows r = (f*Wf + wf)*C + c, M = 24960; N = 128 (freq along H); K = 128
  const float* COS = ws + WS_COS;
  const float* SIN = ws + WS_SIN;
  int mt = blockIdx.x, ntb = blockIdx.y, lane = threadIdx.x;
  int m = lane & 15, hi = lane >> 4;
  int r = mt * 16 + m;
  int c = r % Cq, wf = (r / Cq) % Wf, f = r / (Cq * Wf);
  const float* base = X + (((size_t)f * Hq * Wf + wf) * Cq + c) * 2;  // + kk*390
  int nn0 = ntb * 32 + (lane & 15);
  int nn1 = nn0 + 16;
  v8f rc0 = {}, rs0 = {}, ic0 = {}, is0 = {};
  v8f rc1 = {}, rs1 = {}, ic1 = {}, is1 = {};
  for (int kk = 0; kk < 128; kk += 4) {
    int k0 = kk + hi * 2;
    const float* p0 = base + (size_t)k0 * (Wf * Cq * 2);
    __builtin_prefetch(p0 + 4 * (Wf * Cq * 2), 0, 1);   // global_prefetch_b8
    v2f ar = { p0[0], p0[Wf * Cq * 2] };
    v2f ai = { p0[1], p0[Wf * Cq * 2 + 1] };
    v2f bc0 = { COS[k0 * 128 + nn0], COS[(k0 + 1) * 128 + nn0] };
    v2f bs0 = { SIN[k0 * 128 + nn0], SIN[(k0 + 1) * 128 + nn0] };
    v2f bc1 = { COS[k0 * 128 + nn1], COS[(k0 + 1) * 128 + nn1] };
    v2f bs1 = { SIN[k0 * 128 + nn1], SIN[(k0 + 1) * 128 + nn1] };
    rc0 = wmma_f32x4(ar, bc0, rc0);
    rs0 = wmma_f32x4(ar, bs0, rs0);
    ic0 = wmma_f32x4(ai, bc0, ic0);
    is0 = wmma_f32x4(ai, bs0, is0);
    rc1 = wmma_f32x4(ar, bc1, rc1);
    rs1 = wmma_f32x4(ar, bs1, rs1);
    ic1 = wmma_f32x4(ai, bc1, ic1);
    is1 = wmma_f32x4(ai, bs1, is1);
  }
  for (int vr = 0; vr < 8; ++vr) {
    int rr = mt * 16 + vr + 8 * hi;
    int cc = rr % Cq, wff = (rr / Cq) % Wf, ff = rr / (Cq * Wf);
    size_t rowbase = (((size_t)ff * Hq) * Wf + wff) * Cq + cc;
    {
      float re = (rc0[vr] + sgn * is0[vr]) * scale;
      float im = (ic0[vr] - sgn * rs0[vr]) * scale;
      size_t o = (rowbase + (size_t)nn0 * Wf * Cq) * 2;
      Y[o] = re; Y[o + 1] = im;
    }
    {
      float re = (rc1[vr] + sgn * is1[vr]) * scale;
      float im = (ic1[vr] - sgn * rs1[vr]) * scale;
      size_t o = (rowbase + (size_t)nn1 * Wf * Cq) * 2;
      Y[o] = re; Y[o + 1] = im;
    }
  }
}

// ---------------- log-domain complex scan over T ----------------
struct C2 { float re, im; };
__device__ inline C2 cadd(C2 a, C2 b) { return { a.re + b.re, a.im + b.im }; }
__device__ inline C2 clogc(float x, float y) {
  return { 0.5f * __logf(x * x + y * y + 1e-38f), atan2f(y, x) };
}
__device__ inline C2 clse2(C2 a, C2 b) {
  float m = fmaxf(a.re, b.re);
  float ea = __expf(a.re - m), eb = __expf(b.re - m);
  float sx = ea * __cosf(a.im) + eb * __cosf(b.im);
  float sy = ea * __sinf(a.im) + eb * __sinf(b.im);
  return { m + 0.5f * __logf(sx * sx + sy * sy + 1e-38f), atan2f(sy, sx) };
}
__device__ inline C2 clse3(C2 a, C2 b, C2 c) {
  float m = fmaxf(fmaxf(a.re, b.re), c.re);
  float ea = __expf(a.re - m), eb = __expf(b.re - m), ec = __expf(c.re - m);
  float sx = ea * __cosf(a.im) + eb * __cosf(b.im) + ec * __cosf(c.im);
  float sy = ea * __sinf(a.im) + eb * __sinf(b.im) + ec * __sinf(c.im);
  return { m + 0.5f * __logf(sx * sx + sy * sy + 1e-38f), atan2f(sy, sx) };
}

__global__ void k_scan(float* __restrict__ U, const float* __restrict__ ws) {
  int i = blockIdx.x * blockDim.x + threadIdx.x;   // B*H*Wf*C = 99840
  if (i >= Bq * Hq * Wf * Cq) return;
  int c = i % Cq, wf = (i / Cq) % Wf, h = (i / (Cq * Wf)) % Hq, b = i / (Cq * Wf * Hq);
  int sp = (h * Wf + wf) * Cq + c;
  float ker = ws[WS_KE + 2 * sp], kei = ws[WS_KE + 2 * sp + 1];
  float kir = ws[WS_KI + 2 * sp], kii = ws[WS_KI + 2 * sp + 1];
  const float* gates = ws + WS_GAT;
  C2 Kp[2][2]; C2 Up[2];
  for (int t = 0; t < Tq; ++t) {
    int g = (b * Tq + t) * Cq + c;
    float al = gates[g], de = gates[BT * Cq + g];
    float mu = gates[2 * BT * Cq + g], ga = gates[3 * BT * Cq + g];
    C2 axx = { __logf(0.9f * al), 0.f };
    C2 ayy = { __logf(0.9f * de), 0.f };
    C2 axy = clogc(-kir * mu, -kii * mu);
    C2 ayx = clogc(ker * ga, kei * ga);
    size_t uo = ((((size_t)(b * Tq + t) * Hq + h) * Wf + wf) * Cq + c) * 2;
    C2 lu = clogc(U[uo], U[uo + 1]);
    C2 un = { NEGC, 0.f };
    if (t == 0) {
      Kp[0][0] = axx; Kp[0][1] = axy; Kp[1][0] = ayx; Kp[1][1] = ayy;
      Up[0] = lu; Up[1] = un;
    } else {
      C2 n00 = clse2(cadd(axx, Kp[0][0]), cadd(axy, Kp[1][0]));
      C2 n01 = clse2(cadd(axx, Kp[0][1]), cadd(axy, Kp[1][1]));
      C2 n10 = clse2(cadd(ayx, Kp[0][0]), cadd(ayy, Kp[1][0]));
      C2 n11 = clse2(cadd(ayx, Kp[0][1]), cadd(ayy, Kp[1][1]));
      C2 u0  = clse3(cadd(axx, Up[0]), cadd(axy, Up[1]), lu);
      C2 u1  = clse3(cadd(ayx, Up[0]), cadd(ayy, Up[1]), un);
      Kp[0][0] = n00; Kp[0][1] = n01; Kp[1][0] = n10; Kp[1][1] = n11;
      Up[0] = u0; Up[1] = u1;
    }
    float em = __expf(Up[1].re);                 // Y uses state component 1
    U[uo]     = em * __cosf(Up[1].im);
    U[uo + 1] = em * __sinf(Up[1].im);
  }
}

// ---------------- inverse row rDFT as GEMM (f32 WMMA) ----------------
__global__ __launch_bounds__(32) void k_dft_row_inv(const float* __restrict__ X,
                                                    const float* __restrict__ IB,
                                                    float* __restrict__ out) {
  // rows r = (f*H + h)*C + c, M = 49152; N = 128 (w); K = 130 -> padded 132
  int mt = blockIdx.x, nt = blockIdx.y, lane = threadIdx.x;
  int m = lane & 15, hi = lane >> 4;
  int r = mt * 16 + m;
  int c = r % Cq, h = (r / Cq) & (Hq - 1), f = r / (Cq * Hq);
  const float* row = X + ((size_t)(f * Hq + h) * Wf * Cq + c) * 2;  // kidx -> wf*6+comp
  int nn = nt * 16 + (lane & 15);
  v8f acc = {};
  for (int kk = 0; kk < 132; kk += 4) {
    int k0 = kk + hi * 2;
    v2f a, b;
    for (int j = 0; j < 2; ++j) {
      int ki = k0 + j;
      if (ki < 130) {
        a[j] = row[(ki >> 1) * (Cq * 2) + (ki & 1)];
        b[j] = IB[ki * 128 + nn];
      } else { a[j] = 0.f; b[j] = 0.f; }
    }
    acc = wmma_f32x4(a, b, acc);
  }
  for (int vr = 0; vr < 8; ++vr) {
    int rr = mt * 16 + vr + 8 * hi;
    int cc = rr % Cq, hh = (rr / Cq) & (Hq - 1), ff = rr / (Cq * Hq);
    out[(((size_t)ff * Hq + hh) * Wq + nn) * Cq + cc] = acc[vr];
  }
}

// ---------------- launch ----------------
extern "C" void kernel_launch(void* const* d_in, const int* in_sizes, int n_in,
                              void* d_out, int out_size, void* d_ws, size_t ws_size,
                              hipStream_t stream) {
  const float* input = (const float*)d_in[0];
  const float* convw = (const float*)d_in[1];
  const float* convb = (const float*)d_in[2];
  const float* wa = (const float*)d_in[3];  const float* ba = (const float*)d_in[4];
  const float* wd = (const float*)d_in[5];  const float* bd = (const float*)d_in[6];
  const float* wm = (const float*)d_in[7];  const float* bm = (const float*)d_in[8];
  const float* wg = (const float*)d_in[9];  const float* bg = (const float*)d_in[10];
  const float* kex = (const float*)d_in[11];
  const float* kin = (const float*)d_in[12];
  float* out = (float*)d_out;
  float* ws = (float*)d_ws;
  float* s1 = ws + WS_S1;
  float* u  = ws + WS_U;

  hipMemsetAsync(ws + WS_CTX, 0, 128 * 32 * sizeof(float), stream);

  k_tables<<<(130 * 128 + 255) / 256, 256, 0, stream>>>(ws);
  k_kspec<<<(Hq * Wf * Cq + 255) / 256, 256, 0, stream>>>(kex, kin, ws);
  k_conv<<<dim3(BT, 32), 256, 0, stream>>>(input, convw, convb, ws + WS_CTX);
  k_gates<<<2, 256, 0, stream>>>(ws + WS_CTX, wa, ba, wd, bd, wm, bm, wg, bg, ws + WS_GAT);

  k_dft_row_fwd<<<dim3(3072, 9), 32, 0, stream>>>(input, ws + WS_FB, s1);
  k_dft_col<<<dim3(1560, 4), 32, 0, stream>>>(s1, ws, u, 1.0f, 1.0f);
  k_scan<<<(Bq * Hq * Wf * Cq + 255) / 256, 256, 0, stream>>>(u, ws);
  k_dft_col<<<dim3(1560, 4), 32, 0, stream>>>(u, ws, s1, -1.0f, 1.0f / 128.0f);
  k_dft_row_inv<<<dim3(3072, 8), 32, 0, stream>>>(s1, ws + WS_IB, out);
}
